// MinimalPointPillars_49675591745636
// MI455X (gfx1250) — matrix operand order repaired and checked by reading, not compiled, not run on hardware
//
#include <hip/hip_runtime.h>
#include <stdint.h>

typedef __attribute__((ext_vector_type(16))) _Float16 v16h;
typedef __attribute__((ext_vector_type(8)))  _Float16 v8h;
typedef __attribute__((ext_vector_type(8)))  float    v8f;

#define B_  16
#define P_  2500
#define N_  100
#define HW_ 50

__device__ __forceinline__ v8f wmma_f16(v16h a, v16h b, v8f c) {
  // D = A(16x32 f16) * B(32x16 f16) + C(16x16 f32)
  return __builtin_amdgcn_wmma_f32_16x16x32_f16(false, a, false, b, (short)0, c, false, false);
}

// ---------------------------------------------------------------------------
// Repack fp32 weights [Cout][Cin](...taps) into per-lane B-fragment order.
// B fragment (32x16 f16): col n=lane%16, h=lane/16; K=8h+e (e<8), K=16+8h+(e-8) (e>=8)
// packed idx = ((((t*KC + c)*NT + j)*32 + lane)*16 + e)
// ---------------------------------------------------------------------------
__global__ void repack_weights(const float* __restrict__ w, _Float16* __restrict__ wp,
                               int Cout, int Cin, int Taps, int NT, int KC) {
  int idx = blockIdx.x * 256 + threadIdx.x;
  int total = Taps * KC * NT * 512;
  if (idx >= total) return;
  int e    = idx & 15;
  int lane = (idx >> 4) & 31;
  int rest = idx >> 9;
  int j = rest % NT; rest /= NT;
  int c = rest % KC;
  int t = rest / KC;
  int oc = j * 16 + (lane & 15);
  int k  = ((lane & 16) ? 8 : 0) + (e & 7) + ((e & 8) ? 16 : 0);
  int ci = c * 32 + k;
  float v = (ci < Cin && oc < Cout) ? w[((size_t)oc * Cin + ci) * Taps + t] : 0.0f;
  wp[idx] = (_Float16)v;
}

// ---------------------------------------------------------------------------
// Stage 1: per-point MLP (8->32 relu, 32->32 relu), max over 100 points,
// scatter pillar feature into NHWC BEV canvas [B][50*50][32].
// One wave per pillar; 7 M-tiles of 16 points; WMMA f16 both layers.
// Pillar point-tiles are double-buffered into LDS with the CDNA5 async
// global->LDS engine (global_load_async_to_lds_b128 / s_wait_asynccnt).
// ---------------------------------------------------------------------------
__global__ void __launch_bounds__(128)
pillar_mlp_scatter(const float* __restrict__ pillars, const int* __restrict__ indices,
                   const _Float16* __restrict__ wp1, const float* __restrict__ b1,
                   const _Float16* __restrict__ wp2, const float* __restrict__ b2,
                   float* __restrict__ canvas) {
  __shared__ __align__(16) _Float16 ldsI[4][16 * 32];   // inter-layer tile
  __shared__ __align__(16) float    stg[4][2][128];     // 2 x 512B point staging / wave
  const int lane = threadIdx.x & 31;
  const int wv   = threadIdx.x >> 5;
  const int pid  = blockIdx.x * 4 + wv;   // grid sized exactly: 10000*4 = 40000
  const int b    = pid / P_;
  const int mrow = lane & 15;
  const int c0   = (lane & 16) ? 8 : 0;
  const size_t pbase = (size_t)pid * N_ * 8;

  // weight fragments (packed, one 32B load each)
  v16h W1[2], W2[2];
#pragma unroll
  for (int j = 0; j < 2; ++j) {
    W1[j] = *(const v16h*)(wp1 + ((size_t)(j * 32 + lane)) * 16);
    W2[j] = *(const v16h*)(wp2 + ((size_t)(j * 32 + lane)) * 16);
  }
  v8f Cb1[2], Cb2[2];
#pragma unroll
  for (int j = 0; j < 2; ++j) {
    float c1 = b1[j * 16 + mrow];
    float c2 = b2[j * 16 + mrow];
#pragma unroll
    for (int v = 0; v < 8; ++v) { Cb1[j][v] = c1; Cb2[j][v] = c2; }
  }

  // async staging: each of the 32 lanes copies one 16B chunk -> 512B tile
  const uint32_t stg_base = (uint32_t)(uintptr_t)(&stg[wv][0][0]);
  const int cpt  = lane >> 1;        // point-in-tile covered by this lane
  const int half = (lane & 1) * 4;   // which float4 of the 8-float row

  auto issue_copy = [&](int tile, int buf) {
    int pt = tile * 16 + cpt;
    if (pt > N_ - 1) pt = N_ - 1;    // clamp tail: keeps EXEC uniform, rows masked later
    const float* g = pillars + pbase + (size_t)pt * 8 + half;
    uint32_t loff = stg_base + (uint32_t)buf * 512u + (uint32_t)lane * 16u;
    asm volatile("global_load_async_to_lds_b128 %0, %1, off"
                 :: "v"(loff), "v"(g) : "memory");
  };

  float rmax0 = 0.0f, rmax1 = 0.0f;   // relu outputs are >= 0

  issue_copy(0, 0);
  for (int tI = 0; tI < 7; ++tI) {
    if (tI < 6) {
      issue_copy(tI + 1, (tI + 1) & 1);            // prefetch next tile
      asm volatile("s_wait_asynccnt 1" ::: "memory");  // current tile landed
    } else {
      asm volatile("s_wait_asynccnt 0" ::: "memory");
    }

    // ---- layer-1 A fragment from LDS staging: A[m][k], k = channel (8 real)
    v16h a1;
#pragma unroll
    for (int e = 0; e < 16; ++e) a1[e] = (_Float16)0.0f;
    if (lane < 16) {
      int pt = tI * 16 + lane;
      if (pt < N_) {
        const float4* s4 = (const float4*)(&stg[wv][tI & 1][lane * 8]);
        float4 q0 = s4[0], q1 = s4[1];
        a1[0] = (_Float16)q0.x; a1[1] = (_Float16)q0.y;
        a1[2] = (_Float16)q0.z; a1[3] = (_Float16)q0.w;
        a1[4] = (_Float16)q1.x; a1[5] = (_Float16)q1.y;
        a1[6] = (_Float16)q1.z; a1[7] = (_Float16)q1.w;
      }
    }
    v8f d1a = wmma_f16(a1, W1[0], Cb1[0]);
    v8f d1b = wmma_f16(a1, W1[1], Cb1[1]);

    // relu -> f16 -> LDS as [m][ch] (D layout -> row-major)
#pragma unroll
    for (int v = 0; v < 8; ++v) {
      int m = v + ((lane >> 4) << 3);
      float x0 = d1a[v]; x0 = x0 > 0.0f ? x0 : 0.0f;
      float x1 = d1b[v]; x1 = x1 > 0.0f ? x1 : 0.0f;
      ldsI[wv][m * 32 + mrow]      = (_Float16)x0;
      ldsI[wv][m * 32 + 16 + mrow] = (_Float16)x1;
    }
    __syncthreads();

    // ---- layer-2 A fragment straight from LDS in A-layout
    v8h lo = *(const v8h*)(&ldsI[wv][mrow * 32 + c0]);
    v8h hi = *(const v8h*)(&ldsI[wv][mrow * 32 + c0 + 16]);
    v16h a2;
#pragma unroll
    for (int e = 0; e < 8; ++e) { a2[e] = lo[e]; a2[8 + e] = hi[e]; }

    v8f d2a = wmma_f16(a2, W2[0], Cb2[0]);
    v8f d2b = wmma_f16(a2, W2[1], Cb2[1]);

#pragma unroll
    for (int v = 0; v < 8; ++v) {
      int m = v + ((lane >> 4) << 3);
      if (tI * 16 + m < N_) {   // exclude padded rows from the max
        float x0 = d2a[v]; x0 = x0 > 0.0f ? x0 : 0.0f;
        float x1 = d2b[v]; x1 = x1 > 0.0f ? x1 : 0.0f;
        rmax0 = fmaxf(rmax0, x0);
        rmax1 = fmaxf(rmax1, x1);
      }
    }
    __syncthreads();   // drains DScnt: staging + inter-layer reads done before reuse
  }

  // combine row halves: lane L holds rows 0-7, lane L+16 rows 8-15 (same channel)
  rmax0 = fmaxf(rmax0, __shfl_xor(rmax0, 16, 32));
  rmax1 = fmaxf(rmax1, __shfl_xor(rmax1, 16, 32));

  int xi = indices[pid * 2 + 0]; xi = xi < 0 ? 0 : (xi > HW_ - 1 ? HW_ - 1 : xi);
  int yi = indices[pid * 2 + 1]; yi = yi < 0 ? 0 : (yi > HW_ - 1 ? HW_ - 1 : yi);
  size_t obase = ((size_t)b * (HW_ * HW_) + xi * HW_ + yi) * 32;
  if (lane < 16) {
    canvas[obase + lane]      = rmax0;
    canvas[obase + 16 + lane] = rmax1;
  }
}

// ---------------------------------------------------------------------------
// 3x3 stride-2 pad-1 conv, NHWC fp32 in/out, WMMA f16 with implicit im2col.
// One wave per 16 output pixels x all Cout channels.
// ---------------------------------------------------------------------------
template <int Cin, int Cout, int Hin, int Win, int Hout, int Wout>
__global__ void __launch_bounds__(32)
conv3x3s2_wmma(const float* __restrict__ in, const _Float16* __restrict__ wp,
               const float* __restrict__ bias, float* __restrict__ out) {
  constexpr int NT = Cout / 16;
  constexpr int KC = Cin / 32;
  constexpr int NPIX = Hout * Wout;
  const int lane  = threadIdx.x & 31;
  const int mlane = lane & 15;
  const int c0    = (lane & 16) ? 8 : 0;
  const int bidx  = blockIdx.y;
  const int pix   = blockIdx.x * 16 + mlane;
  const bool vpix = pix < NPIX;
  const int oy = pix / Wout, ox = pix - oy * Wout;

  v8f acc[NT];
#pragma unroll
  for (int j = 0; j < NT; ++j) {
    float bv = bias[j * 16 + mlane];
#pragma unroll
    for (int v = 0; v < 8; ++v) acc[j][v] = bv;
  }

  const float* inb = in + (size_t)bidx * Hin * Win * Cin;
#pragma unroll
  for (int t = 0; t < 9; ++t) {
    int ky = t / 3, kx = t - ky * 3;
    int iy = oy * 2 - 1 + ky;
    int ix = ox * 2 - 1 + kx;
    bool vin = vpix && iy >= 0 && iy < Hin && ix >= 0 && ix < Win;
#pragma unroll
    for (int c = 0; c < KC; ++c) {
      v16h a;
#pragma unroll
      for (int e = 0; e < 16; ++e) a[e] = (_Float16)0.0f;
      if (vin) {
        const float4* s4 =
            (const float4*)(inb + ((size_t)iy * Win + ix) * Cin + c * 32 + c0);
        float4 q0 = s4[0], q1 = s4[1], q2 = s4[4], q3 = s4[5];
        a[0]  = (_Float16)q0.x; a[1]  = (_Float16)q0.y;
        a[2]  = (_Float16)q0.z; a[3]  = (_Float16)q0.w;
        a[4]  = (_Float16)q1.x; a[5]  = (_Float16)q1.y;
        a[6]  = (_Float16)q1.z; a[7]  = (_Float16)q1.w;
        a[8]  = (_Float16)q2.x; a[9]  = (_Float16)q2.y;
        a[10] = (_Float16)q2.z; a[11] = (_Float16)q2.w;
        a[12] = (_Float16)q3.x; a[13] = (_Float16)q3.y;
        a[14] = (_Float16)q3.z; a[15] = (_Float16)q3.w;
      }
#pragma unroll
      for (int j = 0; j < NT; ++j) {
        v16h bfrag =
            *(const v16h*)(wp + ((size_t)((t * KC + c) * NT + j) * 32 + lane) * 16);
        acc[j] = wmma_f16(a, bfrag, acc[j]);
      }
    }
  }

#pragma unroll
  for (int j = 0; j < NT; ++j) {
#pragma unroll
    for (int v = 0; v < 8; ++v) {
      int m = v + ((lane >> 4) << 3);
      int opix = blockIdx.x * 16 + m;
      if (opix < NPIX) {
        float x = acc[j][v]; x = x > 0.0f ? x : 0.0f;
        out[((size_t)bidx * NPIX + opix) * Cout + j * 16 + mlane] = x;
      }
    }
  }
}

// ---------------------------------------------------------------------------
// 1x1 heads: cls (128->3, /2) and box (128->7) -> NCHW flat output
// ---------------------------------------------------------------------------
__global__ void __launch_bounds__(32)
heads_kernel(const float* __restrict__ x, const float* __restrict__ cw,
             const float* __restrict__ cb, const float* __restrict__ bw,
             const float* __restrict__ bb, float* __restrict__ out) {
  const int gid = blockIdx.x;              // b*169 + pix
  const int b = gid / 169, pix = gid - b * 169;
  const int lane = threadIdx.x;
  if (lane >= 10) return;
  const float* xr = x + (size_t)gid * 128;
  const float* wr = (lane < 3) ? (cw + lane * 128) : (bw + (lane - 3) * 128);
  float s = 0.0f;
#pragma unroll 4
  for (int k = 0; k < 128; k += 4)
    s += xr[k] * wr[k] + xr[k + 1] * wr[k + 1] + xr[k + 2] * wr[k + 2] + xr[k + 3] * wr[k + 3];
  if (lane < 3)
    out[((size_t)b * 3 + lane) * 169 + pix] = (s + cb[lane]) * 0.5f;
  else
    out[8112 + ((size_t)b * 7 + (lane - 3)) * 169 + pix] = s + bb[lane - 3];
}

// ---------------------------------------------------------------------------
extern "C" void kernel_launch(void* const* d_in, const int* in_sizes, int n_in,
                              void* d_out, int out_size, void* d_ws, size_t ws_size,
                              hipStream_t stream) {
  (void)in_sizes; (void)n_in; (void)out_size; (void)ws_size;
  const float* pillars = (const float*)d_in[0];
  const int*   indices = (const int*)d_in[1];
  const float* w1  = (const float*)d_in[2];
  const float* b1  = (const float*)d_in[3];
  const float* w2  = (const float*)d_in[4];
  const float* b2  = (const float*)d_in[5];
  const float* c1w = (const float*)d_in[6];
  const float* c1b = (const float*)d_in[7];
  const float* c2w = (const float*)d_in[8];
  const float* c2b = (const float*)d_in[9];
  const float* clw = (const float*)d_in[10];
  const float* clb = (const float*)d_in[11];
  const float* bxw = (const float*)d_in[12];
  const float* bxb = (const float*)d_in[13];
  float* outp = (float*)d_out;

  char* ws = (char*)d_ws;
  float* canvas   = (float*)(ws + 0);          // 16*2500*32 f32 = 5,120,000 B
  float* x1       = (float*)(ws + 5120000);    // 16*625*64  f32 = 2,560,000 B
  float* x2       = (float*)(ws + 7680000);    // 16*169*128 f32 = 1,384,448 B
  _Float16* wp1   = (_Float16*)(ws + 9065472); //  1024 f16
  _Float16* wp2   = (_Float16*)(ws + 9067520); //  1024 f16
  _Float16* wpc1  = (_Float16*)(ws + 9069568); // 18432 f16
  _Float16* wpc2  = (_Float16*)(ws + 9106432); // 73728 f16

  // weight repacks (f32 -> f16 B-fragments)
  repack_weights<<<(1024 + 255) / 256, 256, 0, stream>>>(w1, wp1, 32, 8, 1, 2, 1);
  repack_weights<<<(1024 + 255) / 256, 256, 0, stream>>>(w2, wp2, 32, 32, 1, 2, 1);
  repack_weights<<<(18432 + 255) / 256, 256, 0, stream>>>(c1w, wpc1, 64, 32, 9, 4, 1);
  repack_weights<<<(73728 + 255) / 256, 256, 0, stream>>>(c2w, wpc2, 128, 64, 9, 8, 2);

  hipMemsetAsync(canvas, 0, 5120000, stream);

  // stage 1: MLP + max + scatter (40000 pillars, 4 waves/block, async-pipelined)
  pillar_mlp_scatter<<<10000, 128, 0, stream>>>(pillars, indices, wp1, b1, wp2, b2, canvas);

  // backbone convs
  conv3x3s2_wmma<32, 64, 50, 50, 25, 25><<<dim3(40, 16), 32, 0, stream>>>(canvas, wpc1, c1b, x1);
  conv3x3s2_wmma<64, 128, 25, 25, 13, 13><<<dim3(11, 16), 32, 0, stream>>>(x1, wpc2, c2b, x2);

  // heads
  heads_kernel<<<16 * 169, 32, 0, stream>>>(x2, clw, clb, bxw, bxb, outp);
}
